// FastLinearAttention_9019431322132
// MI455X (gfx1250) — matrix operand order repaired and checked by reading, not compiled, not run on hardware
//
#include <hip/hip_runtime.h>

typedef unsigned short u16;
typedef unsigned int   u32;

typedef __attribute__((ext_vector_type(16))) __bf16 v16bf;
typedef __attribute__((ext_vector_type(8)))  float  v8f;

#define BATCH  4
#define SEQ    8192
#define DIM    1024
#define TRIPLE 3072
#define ROWS   (BATCH * SEQ)   // 32768
#define EPSV   1e-6f

__device__ __forceinline__ u16 f2bf(float f) {
  u32 u = __float_as_uint(f);
  return (u16)((u + 0x7FFFu + ((u >> 16) & 1u)) >> 16);  // RNE
}
__device__ __forceinline__ float bf2f(u16 h) {
  return __uint_as_float(((u32)h) << 16);
}

union FragU { u32 u[8]; v16bf v; };

__device__ __forceinline__ v8f wmma_bf16(v16bf a, v16bf b, v8f c) {
  return __builtin_amdgcn_wmma_f32_16x16x32_bf16(false, a, false, b, (short)0, c, false, false);
}

// A fragment (16x32 bf16). Element (m,k) lives at p[(m0+m)*ld + k], k-pairs contiguous.
// ISA layout: lane L holds row M=L%16; VGPR v holds K = 2*(v&3) + 16*(v>=4) + 8*(L>=16).
__device__ __forceinline__ v16bf frag_a(const u16* p, int m0, int ld, int lane) {
  FragU f;
  const u16* r = p + (size_t)(m0 + (lane & 15)) * ld + ((lane >> 4) << 3);
#pragma unroll
  for (int v = 0; v < 8; ++v)
    f.u[v] = *(const u32*)(r + ((v & 3) << 1) + ((v >> 2) << 4));
  return f.v;
}

// B fragment (32x16 bf16). Element (k,n) lives at p[(n0+n)*ld + k] (K-contiguous storage).
// ISA layout: lane L holds col N=L%16; VGPR v holds K = 2*(v&3) + 16*(v>=4) + 8*(L>=16).
__device__ __forceinline__ v16bf frag_b(const u16* p, int n0, int ld, int lane) {
  FragU f;
  const u16* r = p + (size_t)(n0 + (lane & 15)) * ld + ((lane >> 4) << 3);
#pragma unroll
  for (int v = 0; v < 8; ++v)
    f.u[v] = *(const u32*)(r + ((v & 3) << 1) + ((v >> 2) << 4));
  return f.v;
}

// ---------------- prep kernels ----------------

__global__ __launch_bounds__(256) void convx_kernel(const float* __restrict__ x,
                                                    u16* __restrict__ xb) {
  size_t i = ((size_t)blockIdx.x * 256 + threadIdx.x) * 4;
  float4 v = *(const float4*)(x + i);
  u16 o[4] = { f2bf(v.x), f2bf(v.y), f2bf(v.z), f2bf(v.w) };
  *(u32*)(xb + i)     = *(u32*)&o[0];
  *(u32*)(xb + i + 2) = *(u32*)&o[2];
}

// WcatT[c][d] (K-contiguous, c in [0,3072)):
//  c<1024  : (Wq_head . proj)  (proj folded into Wq, block-diagonal per head)
//  c<2048  : (Wk_head . proj)
//  else    : Wv column
__global__ __launch_bounds__(256) void prepw_kernel(const float* __restrict__ Wq,
                                                    const float* __restrict__ Wk,
                                                    const float* __restrict__ Wv,
                                                    const float* __restrict__ proj,
                                                    u16* __restrict__ WcatT) {
  int idx = blockIdx.x * 256 + threadIdx.x;
  int c = idx >> 10, d = idx & 1023;
  float val;
  if (c < 2048) {
    const float* W = (c < 1024) ? Wq : Wk;
    int cc = c & 1023, h = cc >> 6, fo = cc & 63;
    const float* wrow = W + (size_t)d * 1024 + h * 64;
    float s = 0.f;
#pragma unroll 8
    for (int e = 0; e < 64; ++e) s += wrow[e] * proj[e * 64 + fo];
    val = s;
  } else {
    val = Wv[(size_t)d * 1024 + (c - 2048)];
  }
  WcatT[(size_t)c * 1024 + d] = f2bf(val);
}

__global__ __launch_bounds__(256) void convwo_kernel(const float* __restrict__ Wo,
                                                     u16* __restrict__ WoT) {
  int idx = blockIdx.x * 256 + threadIdx.x;
  int n = idx >> 10, k = idx & 1023;
  WoT[(size_t)n * 1024 + k] = f2bf(Wo[(size_t)k * 1024 + n]);
}

// ---------------- big WMMA GEMM (shared by qkv-GEMM and output-GEMM) ----------------
// C[M][Ncols] = A[M][K](bf16) * Bt[Ncols][K](bf16, K-contiguous)
// 128x128 block tile, BK=32, double-buffered LDS, one barrier per K-step.
// Waves 4(M) x 2(N); each wave 32Mx64N -> 8 WMMA per K-step.
// MODE 0: relu+eps epilogue for cols<2048, bf16 out.  MODE 1: +bias, f32 out.
template <int MODE>
__global__ __launch_bounds__(256)
void gemm_bf16_kernel(const u16* __restrict__ A, const u16* __restrict__ Bt,
                      void* __restrict__ C, const float* __restrict__ bias,
                      int K, int Ncols) {
  __shared__ __align__(16) u16 As[2][128 * 34];  // [m][k], ld 34 -> conflict-free ds
  __shared__ __align__(16) u16 Bs[2][128 * 34];  // [n][k]
  const int tid = threadIdx.x, lane = tid & 31, wave = tid >> 5;
  const int wm = wave >> 1, wn = wave & 1;
  const int m0 = blockIdx.y * 128, n0 = blockIdx.x * 128;

  // staging: 128 rows x 32 halves per matrix; 16 halves (8 dwords) per thread
  const int sr = tid >> 1, sg = (tid & 1) << 4;
  const u32* gA = (const u32*)(A + (size_t)(m0 + sr) * K + sg);
  const u32* gB = (const u32*)(Bt + (size_t)(n0 + sr) * K + sg);
  u32* const sA0 = (u32*)(As[0] + sr * 34 + sg);
  u32* const sB0 = (u32*)(Bs[0] + sr * 34 + sg);
  u32* const sA1 = (u32*)(As[1] + sr * 34 + sg);
  u32* const sB1 = (u32*)(Bs[1] + sr * 34 + sg);

  u32 ra[8], rb[8];
#pragma unroll
  for (int j = 0; j < 8; ++j) { ra[j] = gA[j]; rb[j] = gB[j]; }
#pragma unroll
  for (int j = 0; j < 8; ++j) { sA0[j] = ra[j]; sB0[j] = rb[j]; }
  __syncthreads();

  v8f acc[8] = {};
  int p = 0;
  for (int kb = 0; kb < K; kb += 32) {
    const bool last = (kb + 32 >= K);
    if (!last) {
      const u32* gA2 = gA + ((kb + 32) >> 1);
      const u32* gB2 = gB + ((kb + 32) >> 1);
#pragma unroll
      for (int j = 0; j < 8; ++j) { ra[j] = gA2[j]; rb[j] = gB2[j]; }
      // L2 prefetch (global_prefetch_b8) for the step after next
      __builtin_prefetch(gA2 + 16, 0, 1);
      __builtin_prefetch(gB2 + 16, 0, 1);
    }
    const u16* ap = As[p];
    const u16* bp = Bs[p];
    v16bf a0 = frag_a(ap, wm * 32,      34, lane);
    v16bf a1 = frag_a(ap, wm * 32 + 16, 34, lane);
#pragma unroll
    for (int sn = 0; sn < 4; ++sn) {
      v16bf b = frag_b(bp, wn * 64 + sn * 16, 34, lane);
      acc[sn]     = wmma_bf16(a0, b, acc[sn]);
      acc[4 + sn] = wmma_bf16(a1, b, acc[4 + sn]);
    }
    if (!last) {
      u32* sa = p ? sA0 : sA1;
      u32* sb = p ? sB0 : sB1;
#pragma unroll
      for (int j = 0; j < 8; ++j) { sa[j] = ra[j]; sb[j] = rb[j]; }
    }
    __syncthreads();
    p ^= 1;
  }

#pragma unroll
  for (int t = 0; t < 8; ++t) {
    int sm = t >> 2, sn = t & 3;
    int gm0 = m0 + wm * 32 + sm * 16 + ((lane >> 4) << 3);
    int gn  = n0 + wn * 64 + sn * 16 + (lane & 15);
#pragma unroll
    for (int v = 0; v < 8; ++v) {
      float val = acc[t][v];
      size_t idx = (size_t)(gm0 + v) * Ncols + gn;
      if (MODE == 0) {
        if (gn < 2048) val = fmaxf(val, 0.0f) + EPSV;  // feature map epilogue
        ((u16*)C)[idx] = f2bf(val);
      } else {
        ((float*)C)[idx] = val + bias[gn];
      }
    }
  }
}

// ---------------- kv = kf^T @ v per (b,h): 64x64, K = 8192 ----------------
__global__ __launch_bounds__(256)
void kv_kernel(const u16* __restrict__ qkvf, u16* __restrict__ kvbT) {
  __shared__ __align__(16) u16 kfT[64 * 34];  // [f][n], pairs contiguous in n
  __shared__ __align__(16) u16 vT [64 * 34];  // [d][n]
  const int tid = threadIdx.x, lane = tid & 31, wave = tid >> 5;
  const int bh = blockIdx.x, b = bh >> 4, h = bh & 15;
  const int ft = wave >> 1;                   // f-tile 0..3
  const int dt0 = (wave & 1) << 1, dt1 = dt0 + 1;
  const u16* base = qkvf + (size_t)b * SEQ * TRIPLE + 1024 + h * 64;  // kf; v at +1024
  const int mat = tid >> 7, tt = tid & 127;
  const int row = tt >> 2, f0 = (tt & 3) << 4;
  u16* ldsT = mat ? vT : kfT;
  const u16* gsrc = base + (size_t)mat * 1024 + (size_t)row * TRIPLE + f0;

  v8f acc0 = {}, acc1 = {};
  for (int nb = 0; nb < SEQ; nb += 32) {
    u32 tmp[8];
    const u32* g = (const u32*)(gsrc + (size_t)nb * TRIPLE);
#pragma unroll
    for (int j = 0; j < 8; ++j) tmp[j] = g[j];
    const u16* t16 = (const u16*)tmp;
#pragma unroll
    for (int j = 0; j < 16; ++j) ldsT[(f0 + j) * 34 + row] = t16[j];  // transpose into LDS
    __syncthreads();
    v16bf a  = frag_a(kfT, ft * 16,  34, lane);  // A = kf^T (M=f, K=n)
    v16bf b0 = frag_b(vT,  dt0 * 16, 34, lane);  // B = v    (K=n, N=d)
    v16bf b1 = frag_b(vT,  dt1 * 16, 34, lane);
    acc0 = wmma_bf16(a, b0, acc0);
    acc1 = wmma_bf16(a, b1, acc1);
    __syncthreads();
  }
  int fidx = ft * 16 + ((lane >> 4) << 3);
  int d0 = dt0 * 16 + (lane & 15), d1 = dt1 * 16 + (lane & 15);
  u16* outp = kvbT + (size_t)bh * 4096;  // stored [d][f] (K-contiguous for num GEMM)
#pragma unroll
  for (int v = 0; v < 8; ++v) {
    outp[d0 * 64 + fidx + v] = f2bf(acc0[v]);
    outp[d1 * 64 + fidx + v] = f2bf(acc1[v]);
  }
}

__global__ void ksum_kernel(const u16* __restrict__ qkvf, float* __restrict__ ksum) {
  int bh = blockIdx.x, f = threadIdx.x;
  const u16* p = qkvf + (size_t)(bh >> 4) * SEQ * TRIPLE + 1024 + (bh & 15) * 64 + f;
  float s = 0.f;
  for (int n = 0; n < SEQ; ++n) s += bf2f(p[(size_t)n * TRIPLE]);
  ksum[bh * 64 + f] = s;
}

__global__ __launch_bounds__(256)
void den_kernel(const u16* __restrict__ qkvf, const float* __restrict__ ksum,
                float* __restrict__ den) {
  int idx = blockIdx.x * 256 + threadIdx.x;  // (row, h)
  int row = idx >> 4, h = idx & 15;
  const u16* qf = qkvf + (size_t)row * TRIPLE + h * 64;
  const float* ks = ksum + (size_t)((row >> 13) * 16 + h) * 64;
  float s = 0.f;
#pragma unroll 8
  for (int f = 0; f < 64; ++f) s += bf2f(qf[f]) * ks[f];
  den[idx] = s + EPSV;
}

// ---------------- num = qf @ kv, normalize, pack heads -> attn bf16 ----------------
__global__ __launch_bounds__(256)
void num_kernel(const u16* __restrict__ qkvf, const u16* __restrict__ kvbT,
                const float* __restrict__ den, u16* __restrict__ attn) {
  const int tid = threadIdx.x, lane = tid & 31, wave = tid >> 5;
  const int bh = blockIdx.x, b = bh >> 4, h = bh & 15;
  const int n_loc = blockIdx.y * 128 + wave * 16;
  const size_t row0 = (size_t)b * SEQ + n_loc;
  const u16* kvb = kvbT + (size_t)bh * 4096;

  v8f acc[4] = {};
#pragma unroll
  for (int kb = 0; kb < 64; kb += 32) {
    v16bf a = frag_a(qkvf + row0 * TRIPLE + h * 64 + kb, 0, TRIPLE, lane);
#pragma unroll
    for (int dt = 0; dt < 4; ++dt) {
      v16bf bb = frag_b(kvb + kb, dt * 16, 64, lane);
      acc[dt] = wmma_bf16(a, bb, acc[dt]);
    }
  }
#pragma unroll
  for (int dt = 0; dt < 4; ++dt) {
    int d = dt * 16 + (lane & 15);
    int m0 = (lane >> 4) << 3;
#pragma unroll
    for (int v = 0; v < 8; ++v) {
      size_t row = row0 + m0 + v;
      float val = acc[dt][v] / den[row * 16 + h];
      attn[row * DIM + h * 64 + d] = f2bf(val);
    }
  }
}

extern "C" void kernel_launch(void* const* d_in, const int* in_sizes, int n_in,
                              void* d_out, int out_size, void* d_ws, size_t ws_size,
                              hipStream_t stream) {
  const float* x    = (const float*)d_in[0];
  const float* Wq   = (const float*)d_in[1];
  const float* Wk   = (const float*)d_in[2];
  const float* Wv   = (const float*)d_in[3];
  const float* proj = (const float*)d_in[4];
  const float* Wo   = (const float*)d_in[5];
  const float* bo   = (const float*)d_in[6];

  char* ws = (char*)d_ws;
  size_t off = 0;
  auto alloc = [&](size_t bytes) {
    void* p = ws + off;
    off = (off + bytes + 255) & ~(size_t)255;
    return p;
  };
  u16* xb     = (u16*)alloc((size_t)ROWS * DIM * 2);      // 64 MiB (reused as attn)
  u16* WcatT  = (u16*)alloc((size_t)TRIPLE * DIM * 2);    // 6 MiB
  u16* WoT    = (u16*)alloc((size_t)DIM * DIM * 2);       // 2 MiB
  u16* qkvf   = (u16*)alloc((size_t)ROWS * TRIPLE * 2);   // 192 MiB
  u16* kvbT   = (u16*)alloc((size_t)64 * 4096 * 2);       // 512 KiB
  float* ksum = (float*)alloc((size_t)64 * 64 * 4);
  float* den  = (float*)alloc((size_t)ROWS * 16 * 4);     // 2 MiB
  u16* attn = xb;  // alias: xb dead after GEMM1, same size as attn

  convx_kernel<<<ROWS * DIM / (256 * 4), 256, 0, stream>>>(x, xb);
  prepw_kernel<<<TRIPLE * DIM / 256, 256, 0, stream>>>(Wq, Wk, Wv, proj, WcatT);
  convwo_kernel<<<DIM * DIM / 256, 256, 0, stream>>>(Wo, WoT);
  // qf | kf | v  =  relu(x @ [Wq'|Wk'])+eps , x @ Wv   (206 GFLOP)
  gemm_bf16_kernel<0><<<dim3(TRIPLE / 128, ROWS / 128), 256, 0, stream>>>(
      xb, WcatT, qkvf, nullptr, DIM, TRIPLE);
  ksum_kernel<<<64, 64, 0, stream>>>(qkvf, ksum);
  kv_kernel<<<64, 256, 0, stream>>>(qkvf, kvbT);
  den_kernel<<<ROWS * 16 / 256, 256, 0, stream>>>(qkvf, ksum, den);
  num_kernel<<<dim3(64, SEQ / 128), 256, 0, stream>>>(qkvf, kvbT, den, attn);
  // out = attn @ Wo + bo  (69 GFLOP), f32 output
  gemm_bf16_kernel<1><<<dim3(DIM / 128, ROWS / 128), 256, 0, stream>>>(
      attn, WoT, d_out, bo, DIM, DIM);
}
